// Loss3_26044681683443
// MI455X (gfx1250) — matrix operand order repaired and checked by reading, not compiled, not run on hardware
//
#include <hip/hip_runtime.h>

typedef __attribute__((ext_vector_type(16))) _Float16 v16h;
typedef __attribute__((ext_vector_type(8)))  _Float16 v8h;
typedef __attribute__((ext_vector_type(8)))  float    v8f;

#define S_LEN 128
#define WPB   8   // waves per block (256 threads, wave32)

__global__ void loss3_init(float* acc) {
    acc[0] = 0.0f;
    acc[1] = 0.0f;
}

__global__ __launch_bounds__(256) void loss3_rows(const float* __restrict__ outp,
                                                  const float* __restrict__ trup,
                                                  float* __restrict__ acc,
                                                  int N) {
    __shared__ float sOut[WPB][S_LEN];                      // f32 for precision-sensitive terms
    __shared__ float sTru[WPB][S_LEN];
    __shared__ __align__(16) _Float16 sOutH[WPB][S_LEN];    // f16 for WMMA A gathers (masked base)
    __shared__ __align__(16) _Float16 sTruH[WPB][S_LEN+32]; // f16 duplicated -> mask-free B gathers

    const int lane = threadIdx.x & 31;
    const int w    = threadIdx.x >> 5;
    const int row  = blockIdx.x * WPB + w;
    const bool active = (row < N);   // uniform per wave -> EXEC all-1s inside

    float sq = 0.0f;                 // sum (out-truth)^2 for this row
    float po = -3.402823466e38f;     // output peak
    float pt = -3.402823466e38f;     // truth peak

    if (active) {
        const float* o = outp + (size_t)row * S_LEN;
        const float* t = trup + (size_t)row * S_LEN;
        #pragma unroll
        for (int k = 0; k < 4; ++k) {
            int j = lane + 32 * k;
            float ov = o[j], tv = t[j];
            sOut[w][j] = ov;
            sTru[w][j] = tv;
            _Float16 oh = (_Float16)ov, th = (_Float16)tv;
            sOutH[w][j] = oh;
            sTruH[w][j] = th;
            if (k == 0) sTruH[w][j + S_LEN] = th;   // duplicate first 32 for wrap-free runs
            float d = ov - tv;
            sq += d * d;
            po = fmaxf(po, ov);
            pt = fmaxf(pt, tv);
        }
    }
    __syncthreads();

    if (active) {
        const float*    so  = sOut[w];
        const float*    st  = sTru[w];
        const _Float16* soH = sOutH[w];
        const _Float16* stH = sTruH[w];
        const int m = lane & 15;   // A row (coarse shift) and B column (fine shift)
        const int h = lane >> 4;

        // corr[(16a+b)%128] = sum_j out[(j-16a)%128] * truth[(j+b)%128]
        // 16x16x128 GEMM as 4 chained V_WMMA_F32_16X16X32_F16.
        // A per-lane: two runs of 8 consecutive f16, base multiple of 8 -> aligned b128 loads.
        // B per-lane: run of 16 consecutive f16 at base 32q+16h+m (<=142), mask-free.
        v8f c = {};
        #pragma unroll
        for (int q = 0; q < 4; ++q) {
            int baseA0 = (32 * q +     16 * h - 16 * m) & (S_LEN - 1);
            int baseA1 = (32 * q + 8 + 16 * h - 16 * m) & (S_LEN - 1);
            v8h a0 = *(const v8h*)(soH + baseA0);
            v8h a1 = *(const v8h*)(soH + baseA1);
            v16h af = __builtin_shufflevector(a0, a1, 0, 1, 2, 3, 4, 5, 6, 7,
                                                      8, 9, 10, 11, 12, 13, 14, 15);
            int baseB = 32 * q + 16 * h + m;
            v16h bf;
            #pragma unroll
            for (int i = 0; i < 16; ++i)
                bf[i] = stH[baseB + i];
            c = __builtin_amdgcn_wmma_f32_16x16x32_f16(
                    /*neg_a=*/false, af, /*neg_b=*/false, bf,
                    /*c_mod=*/(short)0, c, /*reuse_a=*/false, /*reuse_b=*/false);
        }

        // argmax corr  <=>  argmin shiftDiff ; tie-break: smallest shift.
        // D VGPR r, lane L  ->  s = 16*r + (L%16)   (both lane halves identical)
        float bc = c[0];
        int   bs = m;
        #pragma unroll
        for (int r = 1; r < 8; ++r) {
            float cv = c[r];
            int   sv = 16 * r + m;
            if (cv > bc) { bc = cv; bs = sv; }
        }
        #pragma unroll
        for (int off = 16; off > 0; off >>= 1) {
            float oc = __shfl_xor(bc, off, 32);
            int   os = __shfl_xor(bs, off, 32);
            if (oc > bc || (oc == bc && os < bs)) { bc = oc; bs = os; }
        }

        // wave reductions for sq / peaks
        #pragma unroll
        for (int off = 16; off > 0; off >>= 1) {
            sq += __shfl_xor(sq, off, 32);
            po  = fmaxf(po, __shfl_xor(po, off, 32));
            pt  = fmaxf(pt, __shfl_xor(pt, off, 32));
        }

        // rolled peak == row peak (roll is a permutation), so scale uses po directly
        const float scale = pt / po;

        // lossDetail: g = cgrad(rolled*scale) - cgrad(truth), rowwise L2 (f32 path)
        float gsq = 0.0f;
        #pragma unroll
        for (int k = 0; k < 4; ++k) {
            int j = lane + 32 * k;
            float sc  = so[(j     - bs) & (S_LEN - 1)] * scale;
            float scm = so[(j - 1 - bs) & (S_LEN - 1)] * scale;
            float scp = so[(j + 1 - bs) & (S_LEN - 1)] * scale;
            float tc  = st[j];
            float tm  = st[(j - 1) & (S_LEN - 1)];
            float tp  = st[(j + 1) & (S_LEN - 1)];
            float g = (sc - 0.5f * (scm + scp)) - (tc - 0.5f * (tm + tp));
            gsq += g * g;
        }
        #pragma unroll
        for (int off = 16; off > 0; off >>= 1)
            gsq += __shfl_xor(gsq, off, 32);

        if (lane == 0) {
            float partial = 50.0f * (float)bs
                          + 100.0f * fabsf(scale - 1.0f)
                          + 0.1f * sqrtf(gsq);
            atomicAdd(&acc[0], partial);   // terms that sum linearly over rows
            atomicAdd(&acc[1], sq);        // MSE needs sqrt AFTER global sum
        }
    }
}

__global__ void loss3_finalize(const float* __restrict__ acc, float* __restrict__ out) {
    out[0] = acc[0] + sqrtf(acc[1]);
}

extern "C" void kernel_launch(void* const* d_in, const int* in_sizes, int n_in,
                              void* d_out, int out_size, void* d_ws, size_t ws_size,
                              hipStream_t stream) {
    const float* output = (const float*)d_in[0];
    const float* truth  = (const float*)d_in[1];
    float* ws  = (float*)d_ws;
    float* out = (float*)d_out;

    const int N = in_sizes[0] / S_LEN;           // 4096
    const int blocks = (N + WPB - 1) / WPB;      // 512 blocks of 8 wave32 rows

    loss3_init<<<1, 1, 0, stream>>>(ws);
    loss3_rows<<<blocks, 32 * WPB, 0, stream>>>(output, truth, ws, N);
    loss3_finalize<<<1, 1, 0, stream>>>(ws, out);
}
